// LongSeqClassifier_20186346291908
// MI455X (gfx1250) — matrix operand order repaired and checked by reading, not compiled, not run on hardware
//
#include <hip/hip_runtime.h>
#include <hip/hip_bf16.h>

typedef __bf16 bf16;
typedef __attribute__((ext_vector_type(16))) __bf16 v16bf;
typedef __attribute__((ext_vector_type(8)))  float  v8f;
typedef int v4i __attribute__((vector_size(16)));
typedef __attribute__((address_space(1))) v4i* as1_v4i;
typedef __attribute__((address_space(3))) v4i* as3_v4i;

#define WMMA_BF16(a, b, c) \
  __builtin_amdgcn_wmma_f32_16x16x32_bf16(false, (a), false, (b), (short)0, (c), false, false)

#if __has_builtin(__builtin_amdgcn_global_load_async_to_lds_b128)
#define HAVE_ASYNC_LDS 1
#else
#define HAVE_ASYNC_LDS 0
#endif

#if HAVE_ASYNC_LDS
#if __has_builtin(__builtin_amdgcn_s_wait_asynccnt)
#define WAIT_ASYNC_6() __builtin_amdgcn_s_wait_asynccnt(6)
#define WAIT_ASYNC_0() __builtin_amdgcn_s_wait_asynccnt(0)
#else
#define WAIT_ASYNC_6() asm volatile("s_wait_asynccnt 0x6" ::: "memory")
#define WAIT_ASYNC_0() asm volatile("s_wait_asynccnt 0x0" ::: "memory")
#endif
#else
#define WAIT_ASYNC_6() ((void)0)
#define WAIT_ASYNC_0() ((void)0)
#endif

__device__ __forceinline__ float sigm(float v) { return 1.0f / (1.0f + __expf(-v)); }
// tanh via sigmoid keeps everything on v_exp (co-executes with XDL WMMA)
__device__ __forceinline__ float ftanh(float v) { return 2.0f * sigm(2.0f * v) - 1.0f; }

// ---------------------------------------------------------------------------
// Fragment helpers (layouts per CDNA5 ISA 7.12.2, wave32)
// ---------------------------------------------------------------------------

// A: 16x32 bf16, row-major source (stride in elements, 16B-aligned rows).
__device__ __forceinline__ v16bf load_a_frag(const bf16* src, int stride, int kbase, int lane) {
  const bf16* p = src + (lane & 15) * stride + kbase + ((lane >> 4) << 3);
  union { uint4 q[2]; v16bf v; } u;
  u.q[0] = *(const uint4*)(p);
  u.q[1] = *(const uint4*)(p + 16);
  return u.v;
}

// B: 32x16 bf16 (KxN) from row-major weight W[N][K] (ldk elements).
__device__ __forceinline__ v16bf load_b_frag(const bf16* w, int ldk, int n0, int kbase, int lane) {
  const bf16* p = w + (size_t)(n0 + (lane & 15)) * ldk + kbase + ((lane >> 4) << 3);
  union { uint4 q[2]; v16bf v; } u;
  u.q[0] = *(const uint4*)(p);
  u.q[1] = *(const uint4*)(p + 16);
  return u.v;
}

// C/D: VGPR r -> (M = (lane>=16 ? 8 : 0)+r, N = lane&15)
__device__ __forceinline__ void store_c_frag(float* g, int ldg, int n0, int lane, v8f acc) {
  int n  = n0 + (lane & 15);
  int m0 = (lane >> 4) << 3;
#pragma unroll
  for (int r = 0; r < 8; ++r) g[(m0 + r) * ldg + n] = acc[r];
}

// One K-step of the 16x128 output strip this wave owns:
// batch-load all 8 B fragments (one clause, partial waits), then 8 WMMAs.
__device__ __forceinline__ void kstep(v8f acc[8], const bf16* a_src, int a_stride,
                                      const bf16* w, int ldk, int n0w, int kbase, int lane) {
  v16bf a = load_a_frag(a_src, a_stride, kbase, lane);
  v16bf bb[8];
#pragma unroll
  for (int nt = 0; nt < 8; ++nt)
    bb[nt] = load_b_frag(w, ldk, n0w + nt * 16, kbase, lane);
#pragma unroll
  for (int nt = 0; nt < 8; ++nt)
    acc[nt] = WMMA_BF16(a, bb[nt], acc[nt]);
}

// ---------------------------------------------------------------------------
// fp32 -> bf16 streaming conversion (x and all weights become bf16)
// ---------------------------------------------------------------------------
__global__ void k_f32_to_bf16(const float* __restrict__ src, bf16* __restrict__ dst, int n) {
  int i = blockIdx.x * blockDim.x + threadIdx.x;
  if (i < n) dst[i] = (bf16)src[i];
}

// ---------------------------------------------------------------------------
// Async (double-buffered) staging of one x_t tile: 16 rows x 768 bf16 = 24 KB.
// 1536 16-byte chunks / 256 threads = 6 issues per thread.
// ---------------------------------------------------------------------------
__device__ __forceinline__ void stage_x(const bf16* __restrict__ xbf, bf16* lds_base,
                                        int seq0, int t, int tid) {
#pragma unroll
  for (int it = 0; it < 6; ++it) {
    int c  = tid + it * 256;
    int r  = c / 96;                 // row 0..15
    int cc = c - r * 96;             // 16B chunk within row
    const bf16* g = xbf + ((size_t)(seq0 + r) * 128 + t) * 768 + cc * 8;
    bf16* l = lds_base + r * 784 + cc * 8;
#if HAVE_ASYNC_LDS
    __builtin_amdgcn_global_load_async_to_lds_b128(
        (as1_v4i)(size_t)(const void*)g,
        (as3_v4i)(unsigned)(size_t)(void*)l,
        0, 0);
#else
    *(uint4*)l = *(const uint4*)g;
#endif
  }
}

// ---------------------------------------------------------------------------
// Layer-1 biLSTM, fused input projection + recurrence + masked mean pooling.
// grid = 32 blocks: (16 sequence tiles) x (2 directions); block = 256 thr.
// Each wave computes 128 gate columns: g = [x_t | h] @ [W_ih | W_hh]^T + b.
// x_t tiles stream into LDS via async-to-LDS, double buffered across steps.
// ---------------------------------------------------------------------------
__global__ void __launch_bounds__(256)
k_lstm_layer1(const bf16* __restrict__ xbf,    // (256,128,768) bf16
              const int*  __restrict__ mask,   // (256,128)
              const bf16* __restrict__ wih_f, const bf16* __restrict__ whh_f,
              const float* __restrict__ bias_f,
              const bf16* __restrict__ wih_b, const bf16* __restrict__ whh_b,
              const float* __restrict__ bias_b,
              float* __restrict__ emb)          // (256,512)
{
  __shared__ __align__(16) bf16  xbuf[2][16][784]; // x_t tiles, double buffered
  __shared__ __align__(16) bf16  hbuf[16][272];    // h state, bf16
  __shared__ __align__(16) float gbuf[16][1032];   // gate pre-activations, f32

  const int dir  = blockIdx.x & 1;
  const int seq0 = (blockIdx.x >> 1) * 16;
  const int tid  = threadIdx.x;
  const int lane = tid & 31;
  const int wave = tid >> 5;
  const int n0w  = wave * 128;

  const bf16*  wih  = dir ? wih_b  : wih_f;
  const bf16*  whh  = dir ? whh_b  : whh_f;
  const float* bias = dir ? bias_b : bias_f;

  // bias per gate column owned by this lane (same for all 8 C rows)
  float bv[8];
#pragma unroll
  for (int nt = 0; nt < 8; ++nt) bv[nt] = bias[n0w + nt * 16 + (lane & 15)];

  // elementwise ownership: row = tid>>4, dims d = (tid&15)+16*j
  const int erow = tid >> 4;
  const int ecol = tid & 15;
  float creg[16], hsum[16];
#pragma unroll
  for (int j = 0; j < 16; ++j) { creg[j] = 0.0f; hsum[j] = 0.0f; }
  float cnt = 0.0f;

  for (int i = tid; i < 16 * 272; i += 256) (&hbuf[0][0])[i] = (bf16)0.0f;

  // prologue: stage step 0 into buffer 0
  stage_x(xbf, &xbuf[0][0][0], seq0, dir ? 127 : 0, tid);

  for (int step = 0; step < 128; ++step) {
    const int t   = dir ? (127 - step) : step;
    const int cur = step & 1;

    if (step + 1 < 128) {
      // issue next step's copy, then wait only for the oldest group (current)
      stage_x(xbf, &xbuf[(step + 1) & 1][0][0], seq0, dir ? (127 - (step + 1)) : (step + 1), tid);
      WAIT_ASYNC_6();
    } else {
      WAIT_ASYNC_0();
    }
    __syncthreads();

    v8f acc[8];
#pragma unroll
    for (int nt = 0; nt < 8; ++nt) {
#pragma unroll
      for (int r = 0; r < 8; ++r) acc[nt][r] = bv[nt];
    }

    // input projection: K = 768 (24 WMMA K-steps)
    for (int kt = 0; kt < 24; ++kt)
      kstep(acc, &xbuf[cur][0][0], 784, wih, 768, n0w, kt * 32, lane);
    // recurrence: K = 256 (8 WMMA K-steps)
#pragma unroll
    for (int kt = 0; kt < 8; ++kt)
      kstep(acc, &hbuf[0][0], 272, whh, 256, n0w, kt * 32, lane);

#pragma unroll
    for (int nt = 0; nt < 8; ++nt)
      store_c_frag(&gbuf[0][0], 1032, n0w + nt * 16, lane, acc[nt]);
    __syncthreads();

    // gate nonlinearities, state update, masked pooling accumulation
    float m = (float)mask[(seq0 + erow) * 128 + t];
    cnt += m;
#pragma unroll
    for (int j = 0; j < 16; ++j) {
      int d = ecol + j * 16;
      float gi = gbuf[erow][d];
      float gf = gbuf[erow][256 + d];
      float gg = gbuf[erow][512 + d];
      float go = gbuf[erow][768 + d];
      float c = sigm(gf) * creg[j] + sigm(gi) * ftanh(gg);
      creg[j] = c;
      float h = sigm(go) * ftanh(c);
      hbuf[erow][d] = (bf16)h;
      hsum[j] += m * h;
    }
    __syncthreads();
  }

  float inv = (cnt > 0.0f) ? (1.0f / cnt) : 0.0f;
#pragma unroll
  for (int j = 0; j < 16; ++j) {
    int d = ecol + j * 16;
    emb[(size_t)(seq0 + erow) * 512 + dir * 256 + d] = hsum[j] * inv;
  }
}

// ---------------------------------------------------------------------------
// Layer-2 biLSTM over (8, 32, 512). Only s[:, -1] is needed: forward runs the
// full 32 steps; backward needs exactly ONE step (reverse scan at t=NS-1).
// grid = 2 (one block per direction), M-tile = 16 (rows 8..15 are padding).
// ---------------------------------------------------------------------------
__global__ void __launch_bounds__(256)
k_lstm_layer2(const float* __restrict__ emb,    // (256,512) == (8,32,512)
              const bf16* __restrict__ wih_f, const bf16* __restrict__ whh_f,
              const float* __restrict__ bias_f,
              const bf16* __restrict__ wih_b, const bf16* __restrict__ whh_b,
              const float* __restrict__ bias_b,
              float* __restrict__ fin)           // (8,512)
{
  __shared__ __align__(16) bf16  xbuf[16][528];
  __shared__ __align__(16) bf16  hbuf[16][272];
  __shared__ __align__(16) float gbuf[16][1032];

  const int dir  = blockIdx.x;
  const int tid  = threadIdx.x;
  const int lane = tid & 31;
  const int wave = tid >> 5;
  const int n0w  = wave * 128;

  const bf16*  wih  = dir ? wih_b  : wih_f;
  const bf16*  whh  = dir ? whh_b  : whh_f;
  const float* bias = dir ? bias_b : bias_f;

  float bv[8];
#pragma unroll
  for (int nt = 0; nt < 8; ++nt) bv[nt] = bias[n0w + nt * 16 + (lane & 15)];

  const int erow = tid >> 4;
  const int ecol = tid & 15;
  float creg[16];
#pragma unroll
  for (int j = 0; j < 16; ++j) creg[j] = 0.0f;

  for (int i = tid; i < 16 * 272; i += 256) (&hbuf[0][0])[i] = (bf16)0.0f;
  __syncthreads();

  const int nsteps = dir ? 1 : 32;
  for (int step = 0; step < nsteps; ++step) {
    const int t = dir ? 31 : step;

    // stage emb_t: rows 0..7 valid, rows 8..15 zero padding
    for (int i = tid; i < 16 * 512; i += 256) {
      int r = i >> 9, k = i & 511;
      float v = (r < 8) ? emb[(size_t)(r * 32 + t) * 512 + k] : 0.0f;
      xbuf[r][k] = (bf16)v;
    }
    __syncthreads();

    v8f acc[8];
#pragma unroll
    for (int nt = 0; nt < 8; ++nt) {
#pragma unroll
      for (int r = 0; r < 8; ++r) acc[nt][r] = bv[nt];
    }
    // input projection: K = 512
    for (int kt = 0; kt < 16; ++kt)
      kstep(acc, &xbuf[0][0], 528, wih, 512, n0w, kt * 32, lane);
    // recurrence: K = 256
#pragma unroll
    for (int kt = 0; kt < 8; ++kt)
      kstep(acc, &hbuf[0][0], 272, whh, 256, n0w, kt * 32, lane);

#pragma unroll
    for (int nt = 0; nt < 8; ++nt)
      store_c_frag(&gbuf[0][0], 1032, n0w + nt * 16, lane, acc[nt]);
    __syncthreads();

#pragma unroll
    for (int j = 0; j < 16; ++j) {
      int d = ecol + j * 16;
      float gi = gbuf[erow][d];
      float gf = gbuf[erow][256 + d];
      float gg = gbuf[erow][512 + d];
      float go = gbuf[erow][768 + d];
      float c = sigm(gf) * creg[j] + sigm(gi) * ftanh(gg);
      creg[j] = c;
      float h = sigm(go) * ftanh(c);
      hbuf[erow][d] = (bf16)h;
      if (step == nsteps - 1 && erow < 8)
        fin[erow * 512 + dir * 256 + d] = h;   // f32, no bf16 loss on output
    }
    __syncthreads();
  }
}

// ---------------------------------------------------------------------------
// Classifier: logits(8,4) = final(8,512) @ cls_w^T(512,4) + cls_b
// ---------------------------------------------------------------------------
__global__ void k_classifier(const float* __restrict__ fin, const float* __restrict__ w,
                             const float* __restrict__ b, float* __restrict__ out) {
  int tid = threadIdx.x;
  if (tid >= 32) return;
  int bi = tid >> 2, c = tid & 3;
  float s = b[c];
  for (int k = 0; k < 512; ++k) s += fin[bi * 512 + k] * w[c * 512 + k];
  out[bi * 4 + c] = s;
}

// ---------------------------------------------------------------------------

extern "C" void kernel_launch(void* const* d_in, const int* in_sizes, int n_in,
                              void* d_out, int out_size, void* d_ws, size_t ws_size,
                              hipStream_t stream) {
  const float* x       = (const float*)d_in[0];
  const int*   mask    = (const int*)d_in[1];
  const float* wl_ih_f = (const float*)d_in[2];
  const float* wl_hh_f = (const float*)d_in[3];
  const float* wl_b_f  = (const float*)d_in[4];
  const float* wl_ih_b = (const float*)d_in[5];
  const float* wl_hh_b = (const float*)d_in[6];
  const float* wl_b_b  = (const float*)d_in[7];
  const float* ws_ih_f = (const float*)d_in[8];
  const float* ws_hh_f = (const float*)d_in[9];
  const float* ws_b_f  = (const float*)d_in[10];
  const float* ws_ih_b = (const float*)d_in[11];
  const float* ws_hh_b = (const float*)d_in[12];
  const float* ws_b_b  = (const float*)d_in[13];
  const float* cls_w   = (const float*)d_in[14];
  const float* cls_b   = (const float*)d_in[15];

  // workspace layout: bf16 x, bf16 weights, then f32 emb + final
  const size_t XN = (size_t)256 * 128 * 768;     // 25,165,824 elems
  bf16* xbf       = (bf16*)d_ws;
  bf16* b_wl_ih_f = xbf + XN;
  bf16* b_wl_ih_b = b_wl_ih_f + 1024 * 768;
  bf16* b_wl_hh_f = b_wl_ih_b + 1024 * 768;
  bf16* b_wl_hh_b = b_wl_hh_f + 1024 * 256;
  bf16* b_ws_ih_f = b_wl_hh_b + 1024 * 256;
  bf16* b_ws_ih_b = b_ws_ih_f + 1024 * 512;
  bf16* b_ws_hh_f = b_ws_ih_b + 1024 * 512;
  bf16* b_ws_hh_b = b_ws_hh_f + 1024 * 256;
  const size_t BF16_ELEMS = XN + 2 * (1024 * 768) + 2 * (1024 * 256)
                          + 2 * (1024 * 512) + 2 * (1024 * 256);
  float* emb = (float*)((char*)d_ws + BF16_ELEMS * sizeof(bf16));   // 256*512
  float* fin = emb + 256 * 512;                                     // 8*512

  auto cvt = [&](const float* s, bf16* d, int n) {
    k_f32_to_bf16<<<(n + 255) / 256, 256, 0, stream>>>(s, d, n);
  };
  cvt(x, xbf, (int)XN);
  cvt(wl_ih_f, b_wl_ih_f, 1024 * 768);
  cvt(wl_ih_b, b_wl_ih_b, 1024 * 768);
  cvt(wl_hh_f, b_wl_hh_f, 1024 * 256);
  cvt(wl_hh_b, b_wl_hh_b, 1024 * 256);
  cvt(ws_ih_f, b_ws_ih_f, 1024 * 512);
  cvt(ws_ih_b, b_ws_ih_b, 1024 * 512);
  cvt(ws_hh_f, b_ws_hh_f, 1024 * 256);
  cvt(ws_hh_b, b_ws_hh_b, 1024 * 256);

  k_lstm_layer1<<<32, 256, 0, stream>>>(xbf, mask,
      b_wl_ih_f, b_wl_hh_f, wl_b_f,
      b_wl_ih_b, b_wl_hh_b, wl_b_b, emb);

  k_lstm_layer2<<<2, 256, 0, stream>>>(emb,
      b_ws_ih_f, b_ws_hh_f, ws_b_f,
      b_ws_ih_b, b_ws_hh_b, ws_b_b, fin);

  k_classifier<<<1, 32, 0, stream>>>(fin, cls_w, cls_b, (float*)d_out);
}
